// DecoderBlock_7095285973398
// MI455X (gfx1250) — matrix operand order repaired and checked by reading, not compiled, not run on hardware
//
#include <hip/hip_runtime.h>
#include <hip/hip_bf16.h>
#include <hip/hip_fp16.h>
#include <math.h>
#include <stdint.h>

// Problem constants (match reference)
#define DB_B   4
#define DB_S   2048
#define DB_D   1024
#define DB_H   16
#define DB_HD  64
#define DB_DFF 4096
#define DB_M   (DB_B * DB_S)   // 8192 rows

typedef _Float16 h16_t;
typedef __attribute__((ext_vector_type(16))) _Float16 v16h;
typedef __attribute__((ext_vector_type(8)))  _Float16 v8h;
typedef __attribute__((ext_vector_type(8)))  float    v8f;

union AFrag { v16h v; v8h h8[2]; h16_t h[16]; };

// ---- explicit global-address-space accessors (avoid FLAT fallback) ----
#define GAS __attribute__((address_space(1)))
__device__ __forceinline__ v16h  gld16(const h16_t* p) { return *(const GAS v16h*)(uintptr_t)p; }
__device__ __forceinline__ v8h   gld8 (const h16_t* p) { return *(const GAS v8h*)(uintptr_t)p; }
__device__ __forceinline__ void  gst8 (h16_t* p, v8h v){ *(GAS v8h*)(uintptr_t)p = v; }
__device__ __forceinline__ void  gsth (h16_t* p, h16_t v){ *(GAS h16_t*)(uintptr_t)p = v; }
__device__ __forceinline__ float gldf (const float* p) { return *(const GAS float*)(uintptr_t)p; }
__device__ __forceinline__ void  gstf (float* p, float v){ *(GAS float*)(uintptr_t)p = v; }

__device__ __forceinline__ v8f wmma_fp16(v16h a, v16h b, v8f c) {
  // D = A(16x32 f16) * B(32x16 f16) + C(16x16 f32)
  return __builtin_amdgcn_wmma_f32_16x16x32_f16(false, a, false, b, (short)0, c, false, false);
}

// A fragment (16x32, MxK) from row-major f16 matrix, leading dim lda.
// Lane L: M = L%16 ; elem e -> K = k0+(L>=16?16:0)+e
__device__ __forceinline__ v16h load_a_frag(const h16_t* A, size_t lda, int row0, int k0) {
  const int lane = threadIdx.x & 31;
  const h16_t* p = A + (size_t)(row0 + (lane & 15)) * lda + (size_t)(k0 + (lane >> 4) * 16);
  AFrag f;
  f.h8[0] = gld8(p);
  f.h8[1] = gld8(p + 8);
  return f.v;
}

// B fragment (32x16, KxN) from a TRANSPOSED operand WT stored [N][K] row-major.
// Lane L: N = n0 + L%16 ; element e -> K = k0 + (L>=16?16:0) + e  => contiguous!
__device__ __forceinline__ v16h load_bT_frag(const h16_t* WT, size_t ldk, int k0, int n0) {
  const int lane = threadIdx.x & 31;
  return gld16(WT + (size_t)(n0 + (lane & 15)) * ldk + (size_t)(k0 + (lane >> 4) * 16));
}

// ---------------------------------------------------------------------------
// f32 -> f16 conversions
// ---------------------------------------------------------------------------
__global__ __launch_bounds__(256) void cvt_f32_f16_kernel(const float* __restrict__ in,
                                                          h16_t* __restrict__ out, int n) {
  int i = blockIdx.x * blockDim.x + threadIdx.x;
  if (i < n) gsth(out + i, (h16_t)gldf(in + i));
}

// in: [K][N] f32  ->  out: [N][K] f16   (both coalesced via LDS tile)
__global__ __launch_bounds__(256) void cvt_transpose_kernel(const float* __restrict__ in,
                                                            h16_t* __restrict__ out,
                                                            int K, int N) {
  __shared__ float tile[32][33];
  const int k0 = (int)blockIdx.x * 32, n0 = (int)blockIdx.y * 32;
  const int tx = threadIdx.x & 31, ty = threadIdx.x >> 5;   // 32 x 8
#pragma unroll
  for (int i = 0; i < 32; i += 8)
    tile[ty + i][tx] = gldf(in + (size_t)(k0 + ty + i) * N + n0 + tx);
  __syncthreads();
#pragma unroll
  for (int i = 0; i < 32; i += 8)
    gsth(out + (size_t)(n0 + ty + i) * K + k0 + tx, (h16_t)tile[tx][ty + i]);
}

// ---------------------------------------------------------------------------
// QKV projection: per (b,h): xh[S,64] @ w[64,64] + bias.
// Weights pre-transposed [64][64]; q,k row-major [B,H,S,64]; v transposed [B,H,64,S].
// One wave = one 16-row tile; 24 WMMAs per wave, all B frags contiguous 32B loads.
// ---------------------------------------------------------------------------
__global__ __launch_bounds__(256) void qkv_kernel(const h16_t* __restrict__ x16,
                                                  const h16_t* __restrict__ wqT,
                                                  const h16_t* __restrict__ wkT,
                                                  const h16_t* __restrict__ wvT,
                                                  const float* __restrict__ bq,
                                                  const float* __restrict__ bk,
                                                  const float* __restrict__ bv,
                                                  h16_t* __restrict__ q,
                                                  h16_t* __restrict__ k,
                                                  h16_t* __restrict__ vT) {
  const int wave = (int)((blockIdx.x * blockDim.x + threadIdx.x) >> 5);
  const int st = wave % (DB_S / 16);
  const int bh = wave / (DB_S / 16);
  const int b = bh / DB_H, h = bh % DB_H;
  const int lane = threadIdx.x & 31;
  const int ln = lane & 15, hi = lane >> 4;
  const int row0 = st * 16;

  const h16_t* xrow = x16 + (size_t)b * DB_S * DB_D + (size_t)h * DB_HD;
  const v16h a0 = load_a_frag(xrow, DB_D, row0, 0);
  const v16h a1 = load_a_frag(xrow, DB_D, row0, 32);

  const size_t obase = ((size_t)b * DB_H + h) * DB_S * DB_HD;  // q,k row-major
  const size_t tbase = ((size_t)b * DB_H + h) * DB_HD * DB_S;  // vT [64][S]

  // --- Q ---
#pragma unroll
  for (int nt = 0; nt < 4; ++nt) {
    v8f acc = {};
    acc = wmma_fp16(a0, load_bT_frag(wqT, DB_HD, 0,  nt * 16), acc);
    acc = wmma_fp16(a1, load_bT_frag(wqT, DB_HD, 32, nt * 16), acc);
    const float bias = gldf(bq + nt * 16 + ln);
#pragma unroll
    for (int r = 0; r < 8; ++r)
      gsth(q + obase + (size_t)(row0 + r + hi * 8) * DB_HD + nt * 16 + ln,
           (h16_t)(acc[r] + bias));
  }
  // --- K ---
#pragma unroll
  for (int nt = 0; nt < 4; ++nt) {
    v8f acc = {};
    acc = wmma_fp16(a0, load_bT_frag(wkT, DB_HD, 0,  nt * 16), acc);
    acc = wmma_fp16(a1, load_bT_frag(wkT, DB_HD, 32, nt * 16), acc);
    const float bias = gldf(bk + nt * 16 + ln);
#pragma unroll
    for (int r = 0; r < 8; ++r)
      gsth(k + obase + (size_t)(row0 + r + hi * 8) * DB_HD + nt * 16 + ln,
           (h16_t)(acc[r] + bias));
  }
  // --- V (store transposed; C-layout rows are contiguous per lane -> v8h store) ---
#pragma unroll
  for (int nt = 0; nt < 4; ++nt) {
    v8f acc = {};
    acc = wmma_fp16(a0, load_bT_frag(wvT, DB_HD, 0,  nt * 16), acc);
    acc = wmma_fp16(a1, load_bT_frag(wvT, DB_HD, 32, nt * 16), acc);
    const float bias = gldf(bv + nt * 16 + ln);
    v8h pk;
#pragma unroll
    for (int r = 0; r < 8; ++r) pk[r] = (h16_t)(acc[r] + bias);
    gst8(vT + tbase + (size_t)(nt * 16 + ln) * DB_S + row0 + hi * 8, pk);
  }
}

// ---------------------------------------------------------------------------
// Flash attention (causal + the reference's "tril==0 -> -inf" quirk).
// One wave = 16 query rows of one (b,h). K is [S,64] (B-frag contiguous in d),
// V is transposed [64,S] (B-frag contiguous in keys). Output f16 [B*S, D].
// ---------------------------------------------------------------------------
__global__ __launch_bounds__(256) void attn_kernel(const h16_t* __restrict__ q,
                                                   const h16_t* __restrict__ k,
                                                   const h16_t* __restrict__ vT,
                                                   h16_t* __restrict__ ao) {
  __shared__ __attribute__((aligned(16))) h16_t pbuf[8][16 * 32];

  const int wavein = threadIdx.x >> 5;
  const int wave = (int)((blockIdx.x * blockDim.x + threadIdx.x) >> 5);
  const int st = wave % (DB_S / 16);
  const int bh = wave / (DB_S / 16);
  const int b = bh / DB_H, h = bh % DB_H;
  const int lane = threadIdx.x & 31;
  const int ln = lane & 15, hi = lane >> 4;
  const int i0 = st * 16;

  const size_t head = ((size_t)b * DB_H + h) * DB_S * DB_HD;
  const h16_t* qp = q + head;
  const h16_t* kp = k + head;
  const h16_t* vp = vT + head;   // [64][S]

  const v16h aq0 = load_a_frag(qp, DB_HD, i0, 0);
  const v16h aq1 = load_a_frag(qp, DB_HD, i0, 32);

  v8f acc[4] = {};
  float m[8], l[8];
#pragma unroll
  for (int r = 0; r < 8; ++r) { m[r] = -INFINITY; l[r] = 0.0f; }

  const int nblk = (i0 + 15) / 32 + 1;
  h16_t* pb = pbuf[wavein];

  for (int jb = 0; jb < nblk; ++jb) {
    const int j0 = jb * 32;

    // --- scores (B frag = 16 contiguous halves of a K row) ---
    v8f s0 = {}, s1 = {};
    {
      const h16_t* kr0 = kp + (size_t)(j0 + ln) * DB_HD + hi * 16;
      s0 = wmma_fp16(aq0, gld16(kr0),      s0);
      s0 = wmma_fp16(aq1, gld16(kr0 + 32), s0);
      const h16_t* kr1 = kp + (size_t)(j0 + 16 + ln) * DB_HD + hi * 16;
      s1 = wmma_fp16(aq0, gld16(kr1),      s1);
      s1 = wmma_fp16(aq1, gld16(kr1 + 32), s1);
    }

    // --- scale + causal/quirk mask; online softmax ---
    float s0v[8], s1v[8], mr[8];
#pragma unroll
    for (int r = 0; r < 8; ++r) {
      const int row = i0 + r + hi * 8;
      float t0 = s0[r] * 0.125f;                       // / sqrt(64)
      t0 = ((j0 + ln) > row || t0 == 0.0f) ? -INFINITY : t0;
      float t1 = s1[r] * 0.125f;
      t1 = ((j0 + 16 + ln) > row || t1 == 0.0f) ? -INFINITY : t1;
      s0v[r] = t0; s1v[r] = t1;
      mr[r] = fmaxf(t0, t1);
    }
#pragma unroll
    for (int off = 8; off > 0; off >>= 1) {
#pragma unroll
      for (int r = 0; r < 8; ++r) mr[r] = fmaxf(mr[r], __shfl_xor(mr[r], off, 16));
    }

    float sc[8], p0v[8], p1v[8], rs[8];
#pragma unroll
    for (int r = 0; r < 8; ++r) {
      const float mn = fmaxf(m[r], mr[r]);
      float s = (mn == -INFINITY) ? 1.0f : __expf(m[r] - mn);
      m[r] = mn;
      sc[r] = s;
      p0v[r] = (s0v[r] == -INFINITY || mn == -INFINITY) ? 0.0f : __expf(s0v[r] - mn);
      p1v[r] = (s1v[r] == -INFINITY || mn == -INFINITY) ? 0.0f : __expf(s1v[r] - mn);
      rs[r] = p0v[r] + p1v[r];
    }
#pragma unroll
    for (int off = 8; off > 0; off >>= 1) {
#pragma unroll
      for (int r = 0; r < 8; ++r) rs[r] += __shfl_xor(rs[r], off, 16);
    }
#pragma unroll
    for (int r = 0; r < 8; ++r) {
      l[r] = l[r] * sc[r] + rs[r];
#pragma unroll
      for (int nt = 0; nt < 4; ++nt) acc[nt][r] *= sc[r];
    }

    // --- P tile (C layout) -> LDS row-major 16x32 -> reload as A frag ---
#pragma unroll
    for (int r = 0; r < 8; ++r) {
      const int row = r + hi * 8;
      pb[row * 32 + ln]      = (h16_t)p0v[r];
      pb[row * 32 + 16 + ln] = (h16_t)p1v[r];
    }
    asm volatile("s_wait_dscnt 0x0" ::: "memory");   // same-wave LDS RAW

    v16h ap;
    {
      AFrag f;
      const h16_t* p = pb + ln * 32 + hi * 16;
      f.h8[0] = *(const v8h*)(p);
      f.h8[1] = *(const v8h*)(p + 8);
      ap = f.v;
    }

    // --- O += P @ V : B frag from vT is contiguous in keys ---
#pragma unroll
    for (int nt = 0; nt < 4; ++nt) {
      const v16h bv_ = gld16(vp + (size_t)(nt * 16 + ln) * DB_S + j0 + hi * 16);
      acc[nt] = wmma_fp16(ap, bv_, acc[nt]);
    }
  }

  // --- epilogue: O /= l, store f16 into concat-head layout [B*S, D] ---
  h16_t* orow = ao + ((size_t)b * DB_S + i0) * DB_D + (size_t)h * DB_HD;
#pragma unroll
  for (int nt = 0; nt < 4; ++nt) {
#pragma unroll
    for (int r = 0; r < 8; ++r)
      gsth(orow + (size_t)(r + hi * 8) * DB_D + nt * 16 + ln, (h16_t)(acc[nt][r] / l[r]));
  }
}

// ---------------------------------------------------------------------------
// Generic f16 GEMM with PRE-TRANSPOSED weights: C[M,N] = A[M,K] @ WT[N,K]^T.
// Block = 8 waves stacked in M (256 rows) over one 64-col strip; each wave does
// a 32x64 tile: per k-step 8 WMMA vs 6 contiguous vmem loads. No LDS, no syncs.
// EPI 0: f32 out = acc + bias[col] + resid[row,col]
// EPI 1: f16 out = gelu_exact(acc + bias[col])
// ---------------------------------------------------------------------------
template <int EPI>
__global__ __launch_bounds__(256) void gemm_kernel(const h16_t* __restrict__ A,
                                                   const h16_t* __restrict__ WT,
                                                   const float* __restrict__ bias,
                                                   const float* __restrict__ resid,
                                                   float* __restrict__ Cf,
                                                   h16_t* __restrict__ Ch,
                                                   int M, int N, int K) {
  const int lane = threadIdx.x & 31;
  const int ln = lane & 15, hi = lane >> 4;
  const int wavein = threadIdx.x >> 5;
  const int nb = N / 64;
  const int bm = (int)blockIdx.x / nb;
  const int bn = (int)blockIdx.x % nb;
  const int m0 = bm * 256 + wavein * 32;
  const int n0 = bn * 64;

  v8f acc[2][4] = {};

  for (int k0 = 0; k0 < K; k0 += 32) {
    const v16h a0 = load_a_frag(A, (size_t)K, m0,      k0);
    const v16h a1 = load_a_frag(A, (size_t)K, m0 + 16, k0);
    if (k0 + 32 < K)
      __builtin_prefetch((const void*)(WT + (size_t)(n0 + ln) * K + k0 + 32 + hi * 16), 0, 0);
#pragma unroll
    for (int nt = 0; nt < 4; ++nt) {
      const v16h bf = load_bT_frag(WT, (size_t)K, k0, n0 + nt * 16);
      acc[0][nt] = wmma_fp16(a0, bf, acc[0][nt]);
      acc[1][nt] = wmma_fp16(a1, bf, acc[1][nt]);
    }
  }

#pragma unroll
  for (int nt = 0; nt < 4; ++nt) {
    const float bcol = gldf(bias + n0 + nt * 16 + ln);
#pragma unroll
    for (int mt = 0; mt < 2; ++mt) {
#pragma unroll
      for (int r = 0; r < 8; ++r) {
        const int row = m0 + mt * 16 + r + hi * 8;
        const int col = n0 + nt * 16 + ln;
        const float c = acc[mt][nt][r] + bcol;
        if (EPI == 0) {
          gstf(Cf + (size_t)row * N + col, c + gldf(resid + (size_t)row * N + col));
        } else {
          const float g = 0.5f * c * (1.0f + erff(c * 0.70710678118654752f)); // exact GELU
          gsth(Ch + (size_t)row * N + col, (h16_t)g);
        }
      }
    }
  }
}

// ---------------------------------------------------------------------------
// LayerNorm over D=1024; one block per row. Optionally also emit f16 copy.
// ---------------------------------------------------------------------------
__global__ __launch_bounds__(256) void ln_kernel(const float* __restrict__ in,
                                                 const float* __restrict__ w,
                                                 const float* __restrict__ b,
                                                 float* __restrict__ outf,
                                                 h16_t* __restrict__ outh,
                                                 int write_h) {
  __shared__ float sm[8];
  const int row = (int)blockIdx.x;
  const float* rp = in + (size_t)row * DB_D;
  const int wv = threadIdx.x >> 5;

  float vals[4];
  float s = 0.0f;
#pragma unroll
  for (int i = 0; i < 4; ++i) { vals[i] = gldf(rp + threadIdx.x + i * 256); s += vals[i]; }
#pragma unroll
  for (int off = 16; off > 0; off >>= 1) s += __shfl_xor(s, off, 32);
  if ((threadIdx.x & 31) == 0) sm[wv] = s;
  __syncthreads();
  float tot = 0.0f;
#pragma unroll
  for (int i = 0; i < 8; ++i) tot += sm[i];
  const float mean = tot * (1.0f / DB_D);

  float vs = 0.0f;
#pragma unroll
  for (int i = 0; i < 4; ++i) { const float d = vals[i] - mean; vs += d * d; }
#pragma unroll
  for (int off = 16; off > 0; off >>= 1) vs += __shfl_xor(vs, off, 32);
  __syncthreads();
  if ((threadIdx.x & 31) == 0) sm[wv] = vs;
  __syncthreads();
  float tv = 0.0f;
#pragma unroll
  for (int i = 0; i < 8; ++i) tv += sm[i];
  const float inv = rsqrtf(tv * (1.0f / DB_D) + 1e-5f);

#pragma unroll
  for (int i = 0; i < 4; ++i) {
    const int c = threadIdx.x + i * 256;
    const float y = (vals[i] - mean) * inv * gldf(w + c) + gldf(b + c);
    gstf(outf + (size_t)row * DB_D + c, y);
    if (write_h) gsth(outh + (size_t)row * DB_D + c, (h16_t)y);
  }
}

// ---------------------------------------------------------------------------
// Host launch
// ---------------------------------------------------------------------------
extern "C" void kernel_launch(void* const* d_in, const int* in_sizes, int n_in,
                              void* d_out, int out_size, void* d_ws, size_t ws_size,
                              hipStream_t stream) {
  (void)in_sizes; (void)n_in; (void)out_size; (void)ws_size;

  const float* x    = (const float*)d_in[0];
  const float* wq   = (const float*)d_in[1];
  const float* bq   = (const float*)d_in[2];
  const float* wk   = (const float*)d_in[3];
  const float* bk   = (const float*)d_in[4];
  const float* wv   = (const float*)d_in[5];
  const float* bv   = (const float*)d_in[6];
  const float* wo   = (const float*)d_in[7];
  const float* bo   = (const float*)d_in[8];
  const float* ln1w = (const float*)d_in[9];
  const float* ln1b = (const float*)d_in[10];
  const float* w1   = (const float*)d_in[11];
  const float* b1   = (const float*)d_in[12];
  const float* w2   = (const float*)d_in[13];
  const float* b2   = (const float*)d_in[14];
  const float* ln2w = (const float*)d_in[15];
  const float* ln2b = (const float*)d_in[16];
  float* out = (float*)d_out;

  char* ws = (char*)d_ws;
  // ---- workspace layout (all sizes multiples of 256B) ----
  constexpr size_t SZ_W64  = (size_t)DB_HD * DB_HD * 2;
  constexpr size_t SZ_MD16 = (size_t)DB_M * DB_D * 2;      // 16 MB
  constexpr size_t SZ_MD32 = (size_t)DB_M * DB_D * 4;      // 32 MB
  constexpr size_t OFF_WQ16 = 0;
  constexpr size_t OFF_WK16 = OFF_WQ16 + SZ_W64;
  constexpr size_t OFF_WV16 = OFF_WK16 + SZ_W64;
  constexpr size_t OFF_WO16 = OFF_WV16 + SZ_W64;
  constexpr size_t OFF_W116 = OFF_WO16 + (size_t)DB_D * DB_D * 2;
  constexpr size_t OFF_W216 = OFF_W116 + (size_t)DB_D * DB_DFF * 2;
  constexpr size_t OFF_X16  = OFF_W216 + (size_t)DB_DFF * DB_D * 2;
  constexpr size_t OFF_Q16  = OFF_X16 + SZ_MD16;
  constexpr size_t OFF_K16  = OFF_Q16 + SZ_MD16;
  constexpr size_t OFF_V16  = OFF_K16 + SZ_MD16;
  constexpr size_t OFF_AO16 = OFF_V16 + SZ_MD16;
  constexpr size_t OFF_Y1   = OFF_AO16 + SZ_MD16;
  constexpr size_t OFF_X1   = OFF_Y1 + SZ_MD32;
  // aliases (lifetimes disjoint):
  constexpr size_t OFF_H16  = OFF_X16;   // 64 MB over dead x16/q16/k16/v16
  constexpr size_t OFF_X116 = OFF_AO16;  // ao16 dead after proj GEMM
  constexpr size_t OFF_Y2   = OFF_Y1;    // y1 dead after LN1

  h16_t* wqT  = (h16_t*)(ws + OFF_WQ16);
  h16_t* wkT  = (h16_t*)(ws + OFF_WK16);
  h16_t* wvT  = (h16_t*)(ws + OFF_WV16);
  h16_t* woT  = (h16_t*)(ws + OFF_WO16);
  h16_t* w1T  = (h16_t*)(ws + OFF_W116);
  h16_t* w2T  = (h16_t*)(ws + OFF_W216);
  h16_t* x16  = (h16_t*)(ws + OFF_X16);
  h16_t* q16  = (h16_t*)(ws + OFF_Q16);
  h16_t* k16  = (h16_t*)(ws + OFF_K16);
  h16_t* vT16 = (h16_t*)(ws + OFF_V16);
  h16_t* ao16 = (h16_t*)(ws + OFF_AO16);
  float* y1   = (float*)(ws + OFF_Y1);
  float* x1   = (float*)(ws + OFF_X1);
  h16_t* h16  = (h16_t*)(ws + OFF_H16);
  h16_t* x116 = (h16_t*)(ws + OFF_X116);
  float* y2   = (float*)(ws + OFF_Y2);

  // weights -> transposed f16 [N][K]
  auto cvtT = [&](const float* src, h16_t* dst, int K, int N) {
    cvt_transpose_kernel<<<dim3(K / 32, N / 32), 256, 0, stream>>>(src, dst, K, N);
  };
  cvtT(wq, wqT, DB_HD, DB_HD);
  cvtT(wk, wkT, DB_HD, DB_HD);
  cvtT(wv, wvT, DB_HD, DB_HD);
  cvtT(wo, woT, DB_D, DB_D);
  cvtT(w1, w1T, DB_D, DB_DFF);
  cvtT(w2, w2T, DB_DFF, DB_D);
  cvt_f32_f16_kernel<<<(DB_M * DB_D + 255) / 256, 256, 0, stream>>>(x, x16, DB_M * DB_D);

  // QKV: B*H*(S/16)=8192 waves = 1024 blocks
  qkv_kernel<<<1024, 256, 0, stream>>>(x16, wqT, wkT, wvT, bq, bk, bv, q16, k16, vT16);

  // Attention
  attn_kernel<<<1024, 256, 0, stream>>>(q16, k16, vT16, ao16);

  // Output projection + bias + residual(x): [8192,1024]@[1024,1024]
  gemm_kernel<0><<<(DB_M / 256) * (DB_D / 64), 256, 0, stream>>>(
      ao16, woT, bo, x, y1, nullptr, DB_M, DB_D, DB_D);

  // LN1 -> x1 (f32 residual) + x1 f16 (next GEMM A)
  ln_kernel<<<DB_M, 256, 0, stream>>>(y1, ln1w, ln1b, x1, x116, 1);

  // FFN up + exact GELU: [8192,4096] = [8192,1024]@[1024,4096]
  gemm_kernel<1><<<(DB_M / 256) * (DB_DFF / 64), 256, 0, stream>>>(
      x116, w1T, b1, nullptr, nullptr, h16, DB_M, DB_DFF, DB_D);

  // FFN down + bias + residual(x1): [8192,1024] = [8192,4096]@[4096,1024]
  gemm_kernel<0><<<(DB_M / 256) * (DB_D / 64), 256, 0, stream>>>(
      h16, w2T, b2, x1, y2, nullptr, DB_M, DB_D, DB_DFF);

  // LN2 -> final output (f32)
  ln_kernel<<<DB_M, 256, 0, stream>>>(y2, ln2w, ln2b, out, nullptr, 0);
}
